// MLStruct_89000312308385
// MI455X (gfx1250) — matrix operand order repaired and checked by reading, not compiled, not run on hardware
//
#include <hip/hip_runtime.h>

// ---------------------------------------------------------------------------
// MLStruct graph kernel for MI455X (gfx1250, wave32, WMMA + TDM).
//
// Pipeline (see round-0 analysis for the reference decomposition + roofline):
//   (0) zero N-float scatter target (L2-resident, 200 KB)
//   (1)+(2) edge MLP (1->64->1) via v_wmma_f32_16x16x32_f16 over 16-edge
//           tiles, fp32 atomic scatter into nsf_accum. Weights staged
//           global->LDS once per workgroup with TENSOR_LOAD_TO_LDS (TDM),
//           then hoisted into per-lane VGPR slices and reused for 10 tiles
//           per wave (uniform trip count -> EXEC all-ones at every WMMA).
//   (3) node MLP, same WMMA/TDM scheme -> nsf output + nsf_v
//   (4) one wave per batch row: 32x32 sparse-row intersection via shuffles,
//       gp MLP spread across lanes. Bernoulli term (~1e-7 rel) dropped.
// ---------------------------------------------------------------------------

#define N_NODES 50000
#define DEG     32
#define E_EDGES (N_NODES * DEG)        // 1,600,000
#define BATCH   1024

#define E_TILES        (E_EDGES / 16)  // 100,000
#define EDGE_BLOCKS    1250
#define EDGE_WAVES     (EDGE_BLOCKS * 8)        // 10,000
#define TILES_PER_WAVE (E_TILES / EDGE_WAVES)   // 10 (exact)

typedef __attribute__((ext_vector_type(16))) _Float16     v16h;
typedef __attribute__((ext_vector_type(8)))  float        v8f;
typedef __attribute__((ext_vector_type(4)))  unsigned int u32x4;
typedef __attribute__((ext_vector_type(8)))  int          i32x8;
typedef __attribute__((ext_vector_type(4)))  int          i32x4;

// --- Tensor Data Mover: DMA one 64-float row from global into LDS. ---------
// D# built per CDNA5 ISA:
//   group0: count=1 | lds_addr | global_addr[56:0] | type=2 (bits 127:126)
//   group1: data_size=4B (bits 17:16), tensor_dim0=64 (bits 79:48),
//           tensor_dim1=1 (bits 111:80), tile_dim0=64 (bits 127:112),
//           tile_dim1/2=0 (unused), tensor_dim0_stride=64 (bits 207:160)
//   groups 2/3: zero (tensor <= 2D)
__device__ __forceinline__ void tdm_row64_f32(const float* g, unsigned lds_off) {
  const unsigned long long ga = (unsigned long long)g;
  u32x4 g0;
  g0[0] = 1u;                                              // count=1, user D#
  g0[1] = lds_off;                                         // lds_addr (bytes)
  g0[2] = (unsigned)ga;                                    // global_addr lo
  g0[3] = ((unsigned)(ga >> 32) & 0x01FFFFFFu) | 0x80000000u; // hi | type=2
  i32x8 g1;
  g1[0] = (int)(2u << 16);    // data_size = 2 (4 bytes)
  g1[1] = (int)(64u << 16);   // tensor_dim0 = 64  (bits 79:48 -> w1[31:16])
  g1[2] = (int)(1u << 16);    // tensor_dim0 hi = 0, tensor_dim1 = 1
  g1[3] = (int)(64u << 16);   // tensor_dim1 hi = 0, tile_dim0 = 64
  g1[4] = 0;                  // tile_dim1 = 0, tile_dim2 = 0 (unused)
  g1[5] = 64;                 // tensor_dim0_stride = 64
  g1[6] = 0;
  g1[7] = 0;
  const i32x4 z4 = {0, 0, 0, 0};
#if __clang_major__ >= 23
  const i32x8 z8 = {0, 0, 0, 0, 0, 0, 0, 0};
  __builtin_amdgcn_tensor_load_to_lds(g0, g1, z4, z4, z8, 0);
#else
  __builtin_amdgcn_tensor_load_to_lds(g0, g1, z4, z4, 0);
#endif
}

// Stage w1|b1|w2 (64 floats each) into sw[0..192) via three TDM descriptors.
// Call with all threads; wave 0 issues the DMAs (TDM ignores EXEC, so the
// single-lane branch issues exactly one op per descriptor) and waits on
// TENSORcnt before the workgroup barrier publishes the data.
__device__ __forceinline__ void stage_weights(float* sw,
                                              const float* w1,
                                              const float* b1,
                                              const float* w2) {
  if (threadIdx.x == 0) {
    const unsigned base = (unsigned)(unsigned long long)(void*)sw;
    tdm_row64_f32(w1, base);
    tdm_row64_f32(b1, base + 256);
    tdm_row64_f32(w2, base + 512);
    __builtin_amdgcn_s_wait_tensorcnt(0);
  }
  __syncthreads();
}

// Per-lane weight slices for the WMMA MLP (layouts per CDNA5 ISA 7.12.2):
//   A 16x32 f16 element i: K = i + (i>=8 ? 8:0) + (lane>=16 ? 8:0)
//   B 32x16 f16 element i: K = i + (i>=8 ? 8:0) + (lane>=16 ? 8:0)  (w2 broadcast over N)
struct MlpRegs {
  float w1r[32], b1r[32];
  v16h  bb0, bb1;
};

__device__ __forceinline__ void load_mlp_regs(MlpRegs& r, const float* sw, int lane) {
  const int lhi8  = (lane & 16) ? 8  : 0;
  const int lhi16 = (lane & 16) ? 16 : 0;
#pragma unroll
  for (int i = 0; i < 16; ++i) {
    const int ka = i + ((i >= 8) ? 8 : 0) + lhi8;
    r.w1r[i]      = sw[ka];
    r.w1r[i + 16] = sw[ka + 32];
    r.b1r[i]      = sw[64 + ka];
    r.b1r[i + 16] = sw[64 + ka + 32];
    const int kb = i + ((i >= 8) ? 8 : 0) + lhi8;
    r.bb0[i] = (_Float16)sw[128 + kb];
    r.bb1[i] = (_Float16)sw[128 + kb + 32];
  }
}

// 16 scalar MLPs per wave: two K=32 WMMAs accumulate the 64-dot in fp32.
__device__ __forceinline__ v8f mlp16_wmma(float x, const MlpRegs& r) {
  v16h a0, a1;
#pragma unroll
  for (int i = 0; i < 16; ++i) {
    a0[i] = (_Float16)fmaxf(fmaf(x, r.w1r[i],      r.b1r[i]),      0.0f);
    a1[i] = (_Float16)fmaxf(fmaf(x, r.w1r[i + 16], r.b1r[i + 16]), 0.0f);
  }
  v8f acc = {};
  acc = __builtin_amdgcn_wmma_f32_16x16x32_f16(false, a0, false, r.bb0,
                                               (short)0, acc, false, false);
  acc = __builtin_amdgcn_wmma_f32_16x16x32_f16(false, a1, false, r.bb1,
                                               (short)0, acc, false, false);
  return acc;
}

__global__ void zero_f32_kernel(float* __restrict__ p, int n) {
  int i = blockIdx.x * blockDim.x + threadIdx.x;
  if (i < n) p[i] = 0.0f;
}

// Stage (1)+(2): edge MLP via WMMA, fp32 atomic scatter. 10 tiles per wave.
__global__ void __launch_bounds__(256)
edge_kernel(const float* __restrict__ edge_w,
            const int*   __restrict__ cols,          // neighbors flat, E
            const float* __restrict__ w1, const float* __restrict__ b1,
            const float* __restrict__ w2, const float* __restrict__ b2,
            float* __restrict__ nsf_accum) {
  __shared__ float sw[192];
  stage_weights(sw, w1, b1, w2);

  const int lane = threadIdx.x & 31;
  MlpRegs r;
  load_mlp_regs(r, sw, lane);

  const float bias = b2[0];
  const int wave = (blockIdx.x * blockDim.x + threadIdx.x) >> 5;
  const int m    = lane & 15;

#pragma unroll 2
  for (int t = 0; t < TILES_PER_WAVE; ++t) {
    const int tile = wave + t * EDGE_WAVES;          // uniform trip count
    const int t0   = tile * 16;
    __builtin_prefetch(cols + t0 + m, 0, 0);         // global_prefetch_b8
    const float x = edge_w[t0 + m];                  // lanes L, L+16 mirror
    v8f acc = mlp16_wmma(x, r);
    // D column 0: lane 0 -> rows 0..7, lane 16 -> rows 8..15 (VGPRs 0..7).
    if (m == 0) {
      const int mbase = (lane == 0) ? 0 : 8;
#pragma unroll
      for (int rr = 0; rr < 8; ++rr)
        atomicAdd(&nsf_accum[cols[t0 + mbase + rr]], acc[rr] + bias);
    }
  }
}

// Stage (3): node MLP via WMMA; writes nsf output and nsf_v scratch.
__global__ void __launch_bounds__(256)
node_kernel(const float* __restrict__ nsf_accum,
            const float* __restrict__ w1, const float* __restrict__ b1,
            const float* __restrict__ w2, const float* __restrict__ b2,
            float* __restrict__ nsf_v, float* __restrict__ out_nsf) {
  __shared__ float sw[192];
  stage_weights(sw, w1, b1, w2);

  const int wave = (blockIdx.x * blockDim.x + threadIdx.x) >> 5;
  const int lane = threadIdx.x & 31;
  const int t0   = wave * 16;                        // N multiple of 16
  if (t0 >= N_NODES) return;                         // wave-uniform

  MlpRegs r;
  load_mlp_regs(r, sw, lane);

  const float x = nsf_accum[t0 + (lane & 15)];
  v8f acc = mlp16_wmma(x, r);
  const float bias = b2[0];

  if ((lane & 15) == 0) {
    const int mbase = (lane == 0) ? 0 : 8;
#pragma unroll
    for (int rr = 0; rr < 8; ++rr) {
      const int i = t0 + mbase + rr;
      const float v = acc[rr] + bias;
      nsf_v[i]   = v;
      out_nsf[i] = v;
    }
  }
}

// Stage (4): one wave per batch row; shuffle-based sparse intersection.
__global__ void __launch_bounds__(256)
pair_kernel(const int*   __restrict__ neighbors,
            const int*   __restrict__ src, const int* __restrict__ dst,
            const float* __restrict__ nsf_v,
            const float* __restrict__ gw1, const float* __restrict__ gb1,
            const float* __restrict__ gw2, const float* __restrict__ gb2,
            float* __restrict__ out) {
  const int wave = (blockIdx.x * blockDim.x + threadIdx.x) >> 5;
  const int lane = threadIdx.x & 31;
  if (wave >= BATCH) return;                         // wave-uniform
  const int b = wave;

  const int s = src[b];
  const int d = dst[b];
  const int   id_s = neighbors[s * DEG + lane];
  const int   id_d = neighbors[d * DEG + lane];
  const float v_s  = nsf_v[id_s];
  const float v_d  = nsf_v[id_d];

  float cross = 0.0f, nss = 0.0f, ndd = 0.0f;
#pragma unroll
  for (int k = 0; k < DEG; ++k) {
    const int   os  = __shfl(id_s, k, 32);
    const float ovs = __shfl(v_s,  k, 32);
    const int   od  = __shfl(id_d, k, 32);
    const float ovd = __shfl(v_d,  k, 32);
    cross += (id_s == od) ? v_s * ovd : 0.0f;
    nss   += (id_s == os) ? v_s * ovs : 0.0f;
    ndd   += (id_d == od) ? v_d * ovd : 0.0f;
  }
#pragma unroll
  for (int off = 16; off > 0; off >>= 1) {
    cross += __shfl_xor(cross, off, 32);
    nss   += __shfl_xor(nss,   off, 32);
    ndd   += __shfl_xor(ndd,   off, 32);
  }

  const float ratio = cross / (sqrtf(nss) * sqrtf(ndd));

  // gp MLP: lane j handles hidden units j and j+32.
  const float h0 = fmaxf(fmaf(ratio, gw1[lane],      gb1[lane]),      0.0f);
  const float h1 = fmaxf(fmaf(ratio, gw1[lane + 32], gb1[lane + 32]), 0.0f);
  float part = h0 * gw2[lane] + h1 * gw2[lane + 32];
#pragma unroll
  for (int off = 16; off > 0; off >>= 1) part += __shfl_xor(part, off, 32);

  if (lane == 0) out[b] = part + gb2[0];
}

extern "C" void kernel_launch(void* const* d_in, const int* in_sizes, int n_in,
                              void* d_out, int out_size, void* d_ws, size_t ws_size,
                              hipStream_t stream) {
  (void)in_sizes; (void)n_in; (void)out_size; (void)ws_size;

  const int*   neighbors = (const int*)d_in[0];
  const float* edge_w    = (const float*)d_in[1];
  const int*   src       = (const int*)d_in[2];
  const int*   dst       = (const int*)d_in[3];
  const float* fe_w1 = (const float*)d_in[4];
  const float* fe_b1 = (const float*)d_in[5];
  const float* fe_w2 = (const float*)d_in[6];
  const float* fe_b2 = (const float*)d_in[7];
  const float* fn_w1 = (const float*)d_in[8];
  const float* fn_b1 = (const float*)d_in[9];
  const float* fn_w2 = (const float*)d_in[10];
  const float* fn_b2 = (const float*)d_in[11];
  const float* gp_w1 = (const float*)d_in[12];
  const float* gp_b1 = (const float*)d_in[13];
  const float* gp_w2 = (const float*)d_in[14];
  const float* gp_b2 = (const float*)d_in[15];

  float* out       = (float*)d_out;          // [0..B) out_struct_n, [B..B+N) nsf
  float* nsf_accum = (float*)d_ws;           // N floats
  float* nsf_v     = nsf_accum + N_NODES;    // N floats

  zero_f32_kernel<<<(N_NODES + 255) / 256, 256, 0, stream>>>(nsf_accum, N_NODES);

  edge_kernel<<<EDGE_BLOCKS, 256, 0, stream>>>(edge_w, neighbors,
                                               fe_w1, fe_b1, fe_w2, fe_b2,
                                               nsf_accum);

  {
    const int waves  = N_NODES / 16;                 // 3,125
    const int blocks = (waves * 32 + 255) / 256;     // 391
    node_kernel<<<blocks, 256, 0, stream>>>(nsf_accum,
                                            fn_w1, fn_b1, fn_w2, fn_b2,
                                            nsf_v, out + BATCH);
  }

  pair_kernel<<<(BATCH * 32) / 256, 256, 0, stream>>>(neighbors, src, dst, nsf_v,
                                                      gp_w1, gp_b1, gp_w2, gp_b2,
                                                      out);
}
